// LSTM_64295660421370
// MI455X (gfx1250) — matrix operand order repaired and checked by reading, not compiled, not run on hardware
//
#include <hip/hip_runtime.h>
#include <hip/hip_bf16.h>

typedef __attribute__((ext_vector_type(16))) __bf16 v16bf;
typedef __attribute__((ext_vector_type(8)))  float  v8f;
typedef unsigned int u32;
typedef unsigned int u32x4 __attribute__((ext_vector_type(4)));
typedef int          i32x4 __attribute__((ext_vector_type(4)));
typedef int          i32x8 __attribute__((ext_vector_type(8)));

#define T_STEPS 1000
#define NB      32       // batch
#define HID     512      // hidden size
#define G4H     2048     // 4*H
#define NUM_WG  64       // workgroups in the scan kernel (all resident)
#define TPB     128      // 4 waves of 32
#define JPW     8        // hidden units per workgroup (cols per gate)
#define LSTR    520      // LDS row stride (elements) for HID rows: 1040 B, 16B aligned
#define KH_STEPS 16      // 512 / 32

// Dynamic LDS layout (bytes); each matrix region is 32 rows x 520 bf16 = 33280 B
#define OFF_WHH 0
#define OFF_WIH 33280
#define OFF_H   66560
#define OFF_X0  99840
#define OFF_X1  133120
#define OFF_A   166400                   // 32 x 32 f32 = 4096
#define SMEM_BYTES 170496

// Toolchain detection: amd_gfx1250_TDM.h ships only with the 6-arg builtin toolchain
#if defined(__has_include)
#if __has_include(<hip/amd_detail/amd_gfx1250_TDM.h>)
#define TDM_6ARG 1
#else
#define TDM_6ARG 0
#endif
#else
#define TDM_6ARG 0
#endif

// ---------------------------------------------------------------------------
// Tensor Data Mover: 2-D tile load Global -> LDS, +4 dwords padding per row
// ---------------------------------------------------------------------------
__device__ __forceinline__ unsigned lds_u32(const void* p) {
    // generic LDS address: low 32 bits are the LDS byte offset (ISA 10.2)
    return (unsigned)(unsigned long long)(uintptr_t)p;
}

__device__ __forceinline__ void tdm_load_2d(unsigned lds_byte_off, const void* gsrc,
                                            unsigned rows, unsigned row_elems,
                                            unsigned g_stride_elems, unsigned interval_code) {
    unsigned long long ga = (unsigned long long)(uintptr_t)gsrc;
    u32x4 g0;
    g0[0] = 1u;                                        // count=1, user descriptor
    g0[1] = lds_byte_off;                              // lds_addr
    g0[2] = (unsigned)ga;                              // global_addr[31:0]
    g0[3] = (unsigned)(ga >> 32) | (2u << 30);         // global_addr[56:32] | type=2
    i32x8 g1;
    g1[0] = (int)((1u << 16)                            // data_size = 2 bytes
                | (1u << 20)                            // pad_enable
                | (interval_code << 22)                 // pad_interval (dwords/row)
                | (3u << 25));                          // pad_amount code 3 => 4 dwords
    g1[1] = (int)((row_elems & 0xFFFFu) << 16);         // tensor_dim0[15:0]
    g1[2] = (int)((row_elems >> 16) | ((rows & 0xFFFFu) << 16));   // dim0 hi | dim1 lo
    g1[3] = (int)((rows >> 16) | ((row_elems & 0xFFFFu) << 16));   // dim1 hi | tile_dim0
    g1[4] = (int)(rows & 0xFFFFu);                      // tile_dim1 | tile_dim2=0
    g1[5] = (int)g_stride_elems;                        // tensor_dim0_stride lo32
    g1[6] = 0;
    g1[7] = 0;
    i32x4 z4 = {0, 0, 0, 0};
#if TDM_6ARG
    i32x8 z8 = {0, 0, 0, 0, 0, 0, 0, 0};
    __builtin_amdgcn_tensor_load_to_lds(g0, g1, z4, z4, z8, /*cpol: DEV scope*/ 16);
#else
    __builtin_amdgcn_tensor_load_to_lds(g0, g1, z4, z4, /*cpol: DEV scope*/ 16);
#endif
}

// ---------------------------------------------------------------------------
// Prep kernels
// ---------------------------------------------------------------------------
__global__ void k_transpose_w(const float* __restrict__ src, __bf16* __restrict__ dst,
                              int K, int Kpad, size_t total) {
    size_t idx = (size_t)blockIdx.x * blockDim.x + threadIdx.x;
    if (idx >= total) return;
    int c = (int)(idx / (size_t)Kpad);
    int k = (int)(idx % (size_t)Kpad);
    float v = (k < K) ? src[(size_t)k * G4H + c] : 0.0f;
    dst[idx] = (__bf16)v;
}

__global__ void k_pad_x(const float* __restrict__ x, __bf16* __restrict__ xbf, size_t total) {
    size_t idx = (size_t)blockIdx.x * blockDim.x + threadIdx.x;
    if (idx >= total) return;
    size_t tn = idx >> 8;
    int k = (int)(idx & 255);
    float v = (k < 240) ? x[tn * 240 + k] : 0.0f;
    xbf[idx] = (__bf16)v;
}

__global__ void k_init_layer(const float* __restrict__ h0l, __bf16* __restrict__ hbuf,
                             unsigned* __restrict__ cnt) {
    int idx = blockIdx.x * blockDim.x + threadIdx.x;
    if (idx < NB * HID) hbuf[idx] = (__bf16)h0l[idx];
    if (idx == 0) *cnt = 0u;
}

// ---------------------------------------------------------------------------
// Persistent LSTM scan for one layer. Weights live in LDS (TDM-loaded once);
// h_{t-1} and x_{t+1} are TDM-staged into LDS each step, overlapped with WMMA.
// All WMMA operands are 16B-aligned LDS rows -> ds_load_b128 fragment loads.
// ---------------------------------------------------------------------------
__global__ __launch_bounds__(TPB, 1) void k_lstm_scan(
    const __bf16* __restrict__ xin,     // [T][NB][kin_pad] bf16 (global)
    int kin_pad, int kx_steps, int xstride /*kin_pad+8*/, unsigned x_icode,
    const __bf16* __restrict__ wih_t,   // [G4H][kin_pad] bf16
    const __bf16* __restrict__ whh_t,   // [G4H][HID]     bf16 (layer slice)
    const float*  __restrict__ bias,    // [G4H]
    const float*  __restrict__ c0,      // [NB][HID]
    __bf16* __restrict__ hbuf,          // [2][NB][HID] global double buffer
    __bf16* __restrict__ ys_bf,         // [T][NB][HID] bf16 out or null
    float*  __restrict__ ys_f,          // [T][NB][HID] f32 out or null
    float*  __restrict__ hT, float* __restrict__ cT,
    unsigned* __restrict__ cnt)
{
    extern __shared__ char smem[];
    __bf16* wlds_hh = (__bf16*)(smem + OFF_WHH);
    __bf16* wlds_ih = (__bf16*)(smem + OFF_WIH);
    __bf16* hlds    = (__bf16*)(smem + OFF_H);
    float*  alds    = (float*)(smem + OFF_A);

    const int tid    = threadIdx.x;
    const int wg     = blockIdx.x;
    const int lane   = tid & 31;
    const int wave   = tid >> 5;
    const int lanehi = lane >> 4;
    const int lanelo = lane & 15;
    const int mtile  = (wave & 1) * 16;
    const int ntile  = (wave >> 1) * 16;

    const int c_local = ntile + lanelo;                              // 0..31
    const int cg_col  = ((c_local >> 3) * HID) + wg * JPW + (c_local & 7);
    const float bval  = bias[cg_col];

    // ---- prologue: TDM-load weight slices (4 gate tiles each) and X_0 ----
    if (wave == 0) {
        for (int g = 0; g < 4; ++g) {
            tdm_load_2d(lds_u32(wlds_hh) + (unsigned)(g * 8 * LSTR * 2),
                        whh_t + (size_t)(g * HID + wg * JPW) * HID, 8, HID, HID, 7);
            tdm_load_2d(lds_u32(wlds_ih) + (unsigned)(g * 8 * xstride * 2),
                        wih_t + (size_t)(g * HID + wg * JPW) * kin_pad,
                        8, (unsigned)kin_pad, (unsigned)kin_pad, x_icode);
        }
        tdm_load_2d(lds_u32((__bf16*)(smem + OFF_X0)), xin,
                    NB, (unsigned)kin_pad, (unsigned)kin_pad, x_icode);
        __builtin_amdgcn_s_wait_tensorcnt(0);
    }

    float creg[2];
    {
        int i0 = tid, i1 = tid + TPB;
        creg[0] = c0[(i0 >> 3) * HID + wg * JPW + (i0 & 7)];
        creg[1] = c0[(i1 >> 3) * HID + wg * JPW + (i1 & 7)];
    }
    __syncthreads();

    for (int t = 0; t < T_STEPS; ++t) {
        const __bf16* hprev_g = hbuf + (size_t)(t & 1) * NB * HID;
        __bf16*       hnext_g = hbuf + (size_t)((t + 1) & 1) * NB * HID;
        const __bf16* xcur  = (const __bf16*)(smem + (((t & 1) != 0) ? OFF_X1 : OFF_X0));
        __bf16*       xnext = (__bf16*)      (smem + ((((t + 1) & 1) != 0) ? OFF_X1 : OFF_X0));

        // ---- stage H_t now; prefetch X_{t+1} (TDM, overlapped with x-part) ----
        if (wave == 0) {
            tdm_load_2d(lds_u32(hlds), hprev_g, NB, HID, HID, 7);
            if (t + 1 < T_STEPS)
                tdm_load_2d(lds_u32(xnext),
                            xin + (size_t)(t + 1) * NB * kin_pad,
                            NB, (unsigned)kin_pad, (unsigned)kin_pad, x_icode);
            __builtin_amdgcn_s_wait_tensorcnt(2);   // X_t complete (in-order TDM)
        }
        __syncthreads();

        v8f acc = {};

        // ---- x_t @ W_ih, all operands in LDS, 16B vector loads ----
        for (int ks = 0; ks < kx_steps; ++ks) {
            const int k0 = ks * 32;
            union { v16bf v; u32x4 q[2]; } A, B;
            const __bf16* asrc = xcur + (size_t)(mtile + lanelo) * xstride + k0 + lanehi * 8;
            const __bf16* bsrc = wlds_ih + (size_t)c_local * xstride + k0 + lanehi * 8;
            A.q[0] = *(const u32x4*)(asrc);        // K pairs 0..7
            A.q[1] = *(const u32x4*)(asrc + 16);   // K pairs 16..23
            B.q[0] = *(const u32x4*)(bsrc);        // K 0..7
            B.q[1] = *(const u32x4*)(bsrc + 16);   // K 16..23
            acc = __builtin_amdgcn_wmma_f32_16x16x32_bf16(false, A.v, false, B.v,
                                                          (short)0, acc, false, false);
        }

        // ---- wait for H_t, then h_{t-1} @ W_hh from LDS ----
        if (wave == 0) {
            if (t + 1 < T_STEPS) __builtin_amdgcn_s_wait_tensorcnt(1);
            else                 __builtin_amdgcn_s_wait_tensorcnt(0);
        }
        __syncthreads();

        for (int ks = 0; ks < KH_STEPS; ++ks) {
            const int k0 = ks * 32;
            union { v16bf v; u32x4 q[2]; } A, B;
            const __bf16* asrc = hlds + (size_t)(mtile + lanelo) * LSTR + k0 + lanehi * 8;
            const __bf16* bsrc = wlds_hh + (size_t)c_local * LSTR + k0 + lanehi * 8;
            A.q[0] = *(const u32x4*)(asrc);
            A.q[1] = *(const u32x4*)(asrc + 16);
            B.q[0] = *(const u32x4*)(bsrc);
            B.q[1] = *(const u32x4*)(bsrc + 16);
            acc = __builtin_amdgcn_wmma_f32_16x16x32_bf16(false, A.v, false, B.v,
                                                          (short)0, acc, false, false);
        }

        // C/D layout: VGPR i -> row mtile + i + 8*lanehi, col = c_local
#pragma unroll
        for (int i = 0; i < 8; ++i) {
            int m = mtile + i + lanehi * 8;
            alds[c_local * 32 + m] = acc[i] + bval;
        }
        __syncthreads();

        // ---- gate nonlinearities + state update ----
#pragma unroll
        for (int e = 0; e < 2; ++e) {
            int idx = tid + e * TPB;
            int m = idx >> 3, jj = idx & 7;
            float ai = alds[(0  + jj) * 32 + m];
            float af = alds[(8  + jj) * 32 + m];
            float ag = alds[(16 + jj) * 32 + m];
            float ao = alds[(24 + jj) * 32 + m];
            float ig = 1.0f / (1.0f + __expf(-ai));
            float fg = 1.0f / (1.0f + __expf(-af));
            float gg = tanhf(ag);
            float og = 1.0f / (1.0f + __expf(-ao));
            float c2 = fg * creg[e] + ig * gg;
            float h2 = og * tanhf(c2);
            creg[e] = c2;
            int jglob = wg * JPW + jj;
            hnext_g[m * HID + jglob] = (__bf16)h2;
            if (ys_bf) ys_bf[((size_t)t * NB + m) * HID + jglob] = (__bf16)h2;
            if (ys_f)  ys_f [((size_t)t * NB + m) * HID + jglob] = h2;
            if (t == T_STEPS - 1) {
                hT[m * HID + jglob] = h2;
                cT[m * HID + jglob] = c2;
            }
        }
        __syncthreads();

        // ---- device-wide step barrier ----
        if (t + 1 < T_STEPS) {
            if (tid == 0) {
                __threadfence();
                __hip_atomic_fetch_add(cnt, 1u, __ATOMIC_RELEASE, __HIP_MEMORY_SCOPE_AGENT);
                unsigned target = (unsigned)(t + 1) * NUM_WG;
                while (__hip_atomic_load(cnt, __ATOMIC_ACQUIRE, __HIP_MEMORY_SCOPE_AGENT) < target) {
                    __builtin_amdgcn_s_sleep(1);
                }
                __threadfence();
            }
            __syncthreads();
        }
    }
}

// ---------------------------------------------------------------------------
// Host launcher
// ---------------------------------------------------------------------------
extern "C" void kernel_launch(void* const* d_in, const int* in_sizes, int n_in,
                              void* d_out, int out_size, void* d_ws, size_t ws_size,
                              hipStream_t stream) {
    const float* x    = (const float*)d_in[0];
    const float* h0   = (const float*)d_in[1];
    const float* c0   = (const float*)d_in[2];
    const float* wih0 = (const float*)d_in[3];
    const float* wih1 = (const float*)d_in[4];
    const float* wih2 = (const float*)d_in[5];
    const float* whh  = (const float*)d_in[6];
    const float* b    = (const float*)d_in[7];
    float* out = (float*)d_out;

    char* ws = (char*)d_ws;
    size_t off = 0;
    unsigned* cnt   = (unsigned*)(ws + off);        off += 256;
    __bf16* xbf     = (__bf16*)(ws + off);          off += (size_t)T_STEPS * NB * 256 * 2;
    __bf16* wih0_t  = (__bf16*)(ws + off);          off += (size_t)G4H * 256 * 2;
    __bf16* wih1_t  = (__bf16*)(ws + off);          off += (size_t)G4H * 512 * 2;
    __bf16* wih2_t  = (__bf16*)(ws + off);          off += (size_t)G4H * 512 * 2;
    __bf16* whh_t   = (__bf16*)(ws + off);          off += (size_t)3 * G4H * 512 * 2;
    __bf16* hbuf    = (__bf16*)(ws + off);          off += (size_t)2 * NB * HID * 2;
    __bf16* ys0     = (__bf16*)(ws + off);          off += (size_t)T_STEPS * NB * HID * 2;
    __bf16* ys1     = (__bf16*)(ws + off);          off += (size_t)T_STEPS * NB * HID * 2;
    (void)off; (void)ws_size; (void)in_sizes; (void)n_in; (void)out_size;

    (void)hipFuncSetAttribute((const void*)k_lstm_scan,
                              hipFuncAttributeMaxDynamicSharedMemorySize, SMEM_BYTES);

    // ---- weight / input prep ----
    {
        size_t tot = (size_t)G4H * 256;
        k_transpose_w<<<(unsigned)((tot + 255) / 256), 256, 0, stream>>>(wih0, wih0_t, 240, 256, tot);
    }
    {
        size_t tot = (size_t)G4H * 512;
        k_transpose_w<<<(unsigned)((tot + 255) / 256), 256, 0, stream>>>(wih1, wih1_t, 512, 512, tot);
        k_transpose_w<<<(unsigned)((tot + 255) / 256), 256, 0, stream>>>(wih2, wih2_t, 512, 512, tot);
        for (int l = 0; l < 3; ++l) {
            k_transpose_w<<<(unsigned)((tot + 255) / 256), 256, 0, stream>>>(
                whh + (size_t)l * HID * G4H, whh_t + (size_t)l * G4H * HID, 512, 512, tot);
        }
    }
    {
        size_t tot = (size_t)T_STEPS * NB * 256;
        k_pad_x<<<(unsigned)((tot + 255) / 256), 256, 0, stream>>>(x, xbf, tot);
    }

    float* ys2_f = out;
    float* hT    = out + (size_t)T_STEPS * NB * HID;
    float* cT    = hT + (size_t)3 * NB * HID;

    const int initBlocks = (NB * HID + 255) / 256;

    // ---- layer 0: kin_pad=256 (rows 512B = 128 dwords -> interval code 6),
    //      LDS row stride 256+8 = 264 elements (528 B, 16B aligned) ----
    k_init_layer<<<initBlocks, 256, 0, stream>>>(h0, hbuf, cnt);
    k_lstm_scan<<<NUM_WG, TPB, SMEM_BYTES, stream>>>(
        xbf, 256, 8, 264, 6u, wih0_t, whh_t, b, c0,
        hbuf, ys0, nullptr, hT, cT, cnt);

    // ---- layer 1: kin_pad=512 (rows 1024B = 256 dwords -> interval code 7),
    //      LDS row stride 512+8 = 520 elements ----
    k_init_layer<<<initBlocks, 256, 0, stream>>>(h0 + (size_t)NB * HID, hbuf, cnt);
    k_lstm_scan<<<NUM_WG, TPB, SMEM_BYTES, stream>>>(
        ys0, 512, 16, 520, 7u, wih1_t, whh_t + (size_t)G4H * HID, b + G4H,
        c0 + (size_t)NB * HID, hbuf, ys1, nullptr,
        hT + (size_t)NB * HID, cT + (size_t)NB * HID, cnt);

    // ---- layer 2 ----
    k_init_layer<<<initBlocks, 256, 0, stream>>>(h0 + (size_t)2 * NB * HID, hbuf, cnt);
    k_lstm_scan<<<NUM_WG, TPB, SMEM_BYTES, stream>>>(
        ys1, 512, 16, 520, 7u, wih2_t, whh_t + (size_t)2 * G4H * HID, b + 2 * G4H,
        c0 + (size_t)2 * NB * HID, hbuf, nullptr, ys2_f,
        hT + (size_t)2 * NB * HID, cT + (size_t)2 * NB * HID, cnt);
}
